// Net_20151986552832
// MI455X (gfx1250) — compile-verified, matrix operand order checked by reading, not run on hardware
//
#include <hip/hip_runtime.h>
#include <cmath>

typedef __attribute__((ext_vector_type(16))) _Float16 v16h;
typedef __attribute__((ext_vector_type(8)))  _Float16 v8h;
typedef __attribute__((ext_vector_type(8)))  float    v8f;

#define EPSF  1e-15f
#define MAXN  0.99999f      /* (1 - 1e-5)/sqrt(c), c=1 */
#define ONE7  0.9999999f    /* 1 - 1e-7 */

// ---- fast transcendentals: single v_rcp/v_sqrt/v_exp/v_log TRANS ops,
// ---- no IEEE div/sqrt fixup chains. All call sites are well-conditioned
// ---- (unit-ball data, denominators clamped >= 1e-15).
__device__ __forceinline__ float frcp(float x)  { return __builtin_amdgcn_rcpf(x); }
__device__ __forceinline__ float fsqrt(float x) { return __builtin_amdgcn_sqrtf(x); }
__device__ __forceinline__ float fdivf(float a, float b) { return a * frcp(b); }
__device__ __forceinline__ float fasinh(float x) {        // log(x + sqrt(x^2+1))
  return __logf(x + fsqrt(__builtin_fmaf(x, x, 1.0f)));
}
__device__ __forceinline__ float fsinh(float x) {
  return 0.5f * (__expf(x) - __expf(-x));
}
__device__ __forceinline__ float fatanh(float x) {        // x in [0,1)
  return 0.5f * __logf(fdivf(1.0f + x, 1.0f - x));
}
__device__ __forceinline__ float ftanh(float x) {         // x >= 0
  const float t = __expf(-2.0f * x);
  return fdivf(1.0f - t, 1.0f + t);
}

__device__ __forceinline__ v8f wmma_f16(v16h a, v16h b, v8f c) {
  return __builtin_amdgcn_wmma_f32_16x16x32_f16(false, a, false, b, (short)0, c,
                                                false, false);
}

// sum across the 16 lanes of a half-wave (xor masks stay inside each half)
__device__ __forceinline__ float halfsum16(float v) {
  v += __shfl_xor(v, 1, 32);
  v += __shfl_xor(v, 2, 32);
  v += __shfl_xor(v, 4, 32);
  v += __shfl_xor(v, 8, 32);
  return v;
}

// ---------------------------------------------------------------------------
// Fused Poincare-MLR epilogue (+ optional HReLU). y[] holds s = x.zu on entry,
// transformed in place to the output point on the ball. Returns c*||out||^2.
// Lane nl in [0,16) owns output feature n = t*16+nl of this row.
// ---------------------------------------------------------------------------
template<int NT, bool HRELU>
__device__ __forceinline__ float poincare_row(float (&y)[NT], float cx2row,
                                              const float* __restrict__ coef,
                                              const float* __restrict__ ch,
                                              const float* __restrict__ sh,
                                              int nl) {
  const float rden = frcp(fmaxf(1.0f - cx2row, EPSF));
  const float opc  = 1.0f + cx2row;
  float part = 0.0f;
#pragma unroll
  for (int t = 0; t < NT; ++t) {
    const int n = t * 16 + nl;
    const float num = 2.0f * y[t] * ch[n] - opc * sh[n];
    const float vv  = coef[n] * fasinh(num * rden);
    const float s   = fsinh(vv);
    y[t] = s;
    part += s * s;
  }
  const float S   = halfsum16(part);
  const float inv = frcp(1.0f + fsqrt(1.0f + S));
  float nrm = fsqrt(S) * inv;
  const float pf = fminf(1.0f, fdivf(MAXN, fmaxf(nrm, EPSF)));
  const float sc = inv * pf;
  nrm *= pf;
#pragma unroll
  for (int t = 0; t < NT; ++t) y[t] *= sc;
  if constexpr (!HRELU) {
    return nrm * nrm;
  } else {
    const float nn  = fmaxf(nrm, EPSF);
    const float lam = fdivf(fatanh(fminf(nn, ONE7)), nn);   // logmap0 factor
    float p2 = 0.0f;
#pragma unroll
    for (int t = 0; t < NT; ++t) {
      const float u = fmaxf(lam * y[t], 0.0f);              // relu in tangent space
      y[t] = u;
      p2 += u * u;
    }
    const float M2 = halfsum16(p2);
    const float mm = fmaxf(fsqrt(M2), EPSF);
    const float th = ftanh(mm);
    const float tt = fdivf(th, mm);                         // expmap0 factor
    const float on = tt * fsqrt(M2);
    const float pf2 = fminf(1.0f, fdivf(MAXN, fmaxf(on, EPSF)));
    const float fac = tt * pf2;
#pragma unroll
    for (int t = 0; t < NT; ++t) y[t] *= fac;
    const float on2 = on * pf2;
    return on2 * on2;
  }
}

// ---------------------------------------------------------------------------
// Weight packing: zu columns -> exact per-lane WMMA B-fragment layout (f16),
// plus per-output-column coef = 2*zn, cosh(2r), sinh(2r). Padded cols -> 0.
// (one-shot tiny kernel: keep precise libm here)
// ---------------------------------------------------------------------------
__global__ void pack_layer(const float* __restrict__ z, const float* __restrict__ r,
                           int K, int N, int steps, int ntiles,
                           float* __restrict__ coef, float* __restrict__ ch,
                           float* __restrict__ sh, _Float16* __restrict__ bfrag) {
  const int Npad = ntiles * 16;
  const int entries = steps * ntiles * 512;
  int idx = blockIdx.x * blockDim.x + threadIdx.x;
  if (idx >= entries) return;
  const int step = idx / (ntiles * 512);
  const int rem  = idx % (ntiles * 512);
  const int nt   = rem / 512;
  const int e    = rem % 512;
  const int lane = e / 16;
  const int h    = e % 16;
  const int nl   = lane & 15;
  const int n    = nt * 16 + nl;
  const int c0   = (lane & 16) ? 8 : 0;
  const int kk   = (h < 8) ? (c0 + h) : (c0 + 16 + (h - 8));
  const int k    = step * 32 + kk;
  float val = 0.0f;
  if (k < K && n < N) {
    float ss = 0.0f;
    for (int q = 0; q < K; ++q) { float zz = z[q * N + n]; ss += zz * zz; }
    const float zn = fmaxf(sqrtf(ss), EPSF);
    val = z[k * N + n] / zn;
  }
  bfrag[idx] = (_Float16)val;
  if (idx < Npad) {
    if (idx < N) {
      float ss = 0.0f;
      for (int q = 0; q < K; ++q) { float zz = z[q * N + idx]; ss += zz * zz; }
      const float zn = fmaxf(sqrtf(ss), EPSF);
      const float d  = 2.0f * r[idx];
      coef[idx] = 2.0f * zn;  ch[idx] = coshf(d);  sh[idx] = sinhf(d);
    } else {
      coef[idx] = 0.0f;  ch[idx] = 0.0f;  sh[idx] = 0.0f;   // kill padded cols
    }
  }
}

// logmap0 over channel axis * beta-ratio; NCHW input -> NHWC output
template<int C>
__global__ void logmap_prep(const float* __restrict__ x, float* __restrict__ v,
                            int HW, int total, float ratio) {
  int idx = blockIdx.x * blockDim.x + threadIdx.x;
  if (idx >= total) return;
  const int b = idx / HW, p = idx % HW;
  float vals[C]; float n2 = 0.0f;
#pragma unroll
  for (int c = 0; c < C; ++c) {
    float vv = x[(b * C + c) * HW + p];
    vals[c] = vv; n2 += vv * vv;
  }
  const float nn  = fmaxf(fsqrt(n2), EPSF);
  const float lam = fdivf(fatanh(fminf(nn, ONE7)), nn) * ratio;
#pragma unroll
  for (int c = 0; c < C; ++c) v[(b * HW + p) * C + c] = vals[c] * lam;
}

// ---------------------------------------------------------------------------
// Implicit-GEMM conv with WMMA, fully fused:
//   - raw 5x5xCIN patch gathered once into registers (offsets fold to
//     compile-time constants; one v_cndmask selects between the two
//     half-wave variants -- no div/mod chains, no exec-mask branches)
//   - patch expmap0 factor + c||xp||^2 computed in-wave (lane m and lane m+16
//     jointly hold all K values of row m: one shfl_xor(16) completes the sum)
//   - f16 WMMA accumulate, then fused Poincare-MLR + HReLU epilogue
// ---------------------------------------------------------------------------
template<int CIN, int IW, int OW, int STEPS, int KREAL, int NREAL>
__global__ void conv_gemm(const float* __restrict__ v,
                          const _Float16* __restrict__ bfrag,
                          const float* __restrict__ coef, const float* __restrict__ ch,
                          const float* __restrict__ sh, float* __restrict__ out,
                          int Mtiles) {
  const int wave = blockIdx.x * (blockDim.x >> 5) + (threadIdx.x >> 5);
  if (wave >= Mtiles) return;
  const int lane = threadIdx.x & 31;
  const int m    = lane & 15;
  const bool hiw = (lane & 16) != 0;       // upper half-wave -> c0 = 8
  const int tile = wave * 16;
  const int row  = tile + m;
  const int L = OW * OW;
  const int b = row / L, p = row % L, oy = p / OW, ox = p % OW;
  const float* base = v + (size_t)b * (IW * IW * CIN);
  const int pixbase = (oy * IW + ox) * CIN;

  // patch-element offset for a given k (fully folds after unrolling)
  auto off_of = [](int k) {
    const int cc = k / 25, r2 = k % 25, i = r2 / 5, j = r2 % 5;
    return (i * IW + j) * CIN + cc;
  };

  // --- pass 1: gather raw patch values, accumulate partial ||patch||^2 ---
  float raw[STEPS * 16];
  float s = 0.0f;
#pragma unroll
  for (int step = 0; step < STEPS; ++step) {
#pragma unroll
    for (int h = 0; h < 16; ++h) {
      const int kb = (h < 8) ? h : (16 + h - 8);
      const int k0 = step * 32 + kb;       // compile-time (c0 = 0 variant)
      const int k1 = k0 + 8;               // compile-time (c0 = 8 variant)
      const bool ok0 = (k0 < KREAL), ok1 = (k1 < KREAL);
      float val = 0.0f;
      if (ok0 || ok1) {
        const int o0 = ok0 ? off_of(k0) : 0;
        const int o1 = ok1 ? off_of(k1) : 0;
        const int off = hiw ? o1 : o0;     // single select of two constants
        val = base[pixbase + off];
        if (ok0 != ok1) {                  // boundary element: zero one half
          const bool inb = hiw ? ok1 : ok0;
          val = inb ? val : 0.0f;
        }
      }
      raw[step * 16 + h] = val;
      s += val * val;
    }
  }
  // complete per-row patch norm: lane m + lane m+16 cover all K of row m
  const float stot = s + __shfl_xor(s, 16, 32);
  const float nn = fmaxf(fsqrt(stot), EPSF);
  const float t  = fdivf(ftanh(nn), nn);                 // expmap0 factor
  const float on = t * fsqrt(stot);
  const float pfp = fminf(1.0f, fdivf(MAXN, fmaxf(on, EPSF))); // ball projection
  const float fr  = t * pfp;
  const float cxrow = fr * fr * stot;                    // c*||xp||^2 of row m

  // --- pass 2: scale -> f16 fragments, WMMA accumulate ---
  v8f acc = {};
#pragma unroll
  for (int step = 0; step < STEPS; ++step) {
    v16h a;
#pragma unroll
    for (int h = 0; h < 16; ++h) a[h] = (_Float16)(raw[step * 16 + h] * fr);
    v16h bf = *(const v16h*)(bfrag + step * 512 + lane * 16);
    acc = wmma_f16(a, bf, acc);
  }

  // --- fused epilogue ---
  const int nl = lane & 15, half = lane >> 4;
#pragma unroll
  for (int g = 0; g < 8; ++g) {
    const int orow = tile + half * 8 + g;
    const float cx = __shfl(cxrow, half * 8 + g, 32);   // broadcast row cx2
    float y[1]; y[0] = acc[g];
    poincare_row<1, true>(y, cx, coef, ch, sh, nl);
    if (nl < NREAL) out[(size_t)orow * NREAL + nl] = y[0];
  }
}

// ---------------------------------------------------------------------------
// hyperbolic 2x2 maxpool on NHWC (argmax of channel-norm^2 over the patch),
// optionally fused with the next layer's logmap0 * beta-ratio (the winner's
// squared norm is already in hand).
// ---------------------------------------------------------------------------
template<int C, int IWW, bool LOGMAP>
__global__ void hmaxpool(const float* __restrict__ in, float* __restrict__ out,
                         int total, float ratio) {
  const int OWW = IWW / 2;
  int idx = blockIdx.x * blockDim.x + threadIdx.x;
  if (idx >= total) return;
  const int b = idx / (OWW * OWW), p = idx % (OWW * OWW);
  const int oy = p / OWW, ox = p % OWW;
  float best = -1.0f; int bi = 0;
#pragma unroll
  for (int q = 0; q < 4; ++q) {
    const int dy = q >> 1, dx = q & 1;
    const float* pp = in + (size_t)((b * IWW + 2 * oy + dy) * IWW + 2 * ox + dx) * C;
    float n2 = 0.0f;
#pragma unroll
    for (int c = 0; c < C; ++c) n2 += pp[c] * pp[c];
    if (n2 > best) { best = n2; bi = q; }
  }
  const float* src = in + (size_t)((b * IWW + 2 * oy + (bi >> 1)) * IWW + 2 * ox + (bi & 1)) * C;
  float* dst = out + (size_t)(b * OWW * OWW + p) * C;
  float lam = 1.0f;
  if constexpr (LOGMAP) {
    const float nn = fmaxf(fsqrt(fmaxf(best, 0.0f)), EPSF);
    lam = fdivf(fatanh(fminf(nn, ONE7)), nn) * ratio;
  }
#pragma unroll
  for (int c = 0; c < C; ++c) dst[c] = src[c] * lam;
}

// beta-concatenation flatten: [B,5,5,16] NHWC -> x_fc [B,416] f16 (+ cx2)
__global__ void flatten_k(const float* __restrict__ p2, _Float16* __restrict__ xf,
                          float* __restrict__ cx2f, float ratio) {
  int b = blockIdx.x * blockDim.x + threadIdx.x;
  if (b >= 1024) return;
  const float* base = p2 + (size_t)b * 400;
  float lam[25]; float S = 0.0f;
  for (int p = 0; p < 25; ++p) {
    float n2 = 0.0f;
#pragma unroll
    for (int c = 0; c < 16; ++c) { float vv = base[p * 16 + c]; n2 += vv * vv; }
    const float nn = fmaxf(fsqrt(n2), EPSF);
    const float l  = fdivf(fatanh(fminf(nn, ONE7)), nn) * ratio;
    lam[p] = l;  S += l * l * n2;
  }
  const float nn = fmaxf(fsqrt(S), EPSF);
  const float t  = fdivf(ftanh(nn), nn);
  const float on = t * fsqrt(S);
  const float pf = fminf(1.0f, fdivf(MAXN, fmaxf(on, EPSF)));
  const float fac = t * pf;
  cx2f[b] = (on * pf) * (on * pf);
  _Float16* xr = xf + (size_t)b * 416;
  for (int p = 0; p < 25; ++p)
#pragma unroll
    for (int c = 0; c < 16; ++c)
      xr[c * 25 + p] = (_Float16)(base[p * 16 + c] * lam[p] * fac);   // torch order c*25+p
  for (int q = 400; q < 416; ++q) xr[q] = (_Float16)0.0f;
}

// ---------------------------------------------------------------------------
// Dense Poincare-FC with WMMA: row padded-f16 A, multi n-tile accumulators so
// the full row reduction stays in the wave. Hidden layers emit next-layer f16
// input + cx2; final layer writes f32 logits.
// ---------------------------------------------------------------------------
template<int KPAD, int STEPS, int NTILES, int NREAL, bool FINAL>
__global__ void fc_gemm(const _Float16* __restrict__ A, const float* __restrict__ cx2in,
                        const _Float16* __restrict__ bfrag,
                        const float* __restrict__ coef, const float* __restrict__ ch,
                        const float* __restrict__ sh,
                        _Float16* __restrict__ nxt, float* __restrict__ cx2out,
                        float* __restrict__ fout, int Mtiles) {
  const int wave = blockIdx.x * (blockDim.x >> 5) + (threadIdx.x >> 5);
  if (wave >= Mtiles) return;
  const int lane = threadIdx.x & 31;
  const int m  = lane & 15;
  const int c0 = (lane & 16) ? 8 : 0;
  const int tile = wave * 16;
  const int row  = tile + m;
  v8f zero = {};
  v8f acc[NTILES];
#pragma unroll
  for (int t = 0; t < NTILES; ++t) acc[t] = zero;
#pragma unroll
  for (int step = 0; step < STEPS; ++step) {
    const _Float16* ap = A + (size_t)row * KPAD + step * 32;
    v8h lo = *(const v8h*)(ap + c0);
    v8h hi = *(const v8h*)(ap + c0 + 16);
    v16h a = __builtin_shufflevector(lo, hi, 0,1,2,3,4,5,6,7,8,9,10,11,12,13,14,15);
#pragma unroll
    for (int t = 0; t < NTILES; ++t) {
      v16h bf = *(const v16h*)(bfrag + (size_t)(step * NTILES + t) * 512 + lane * 16);
      acc[t] = wmma_f16(a, bf, acc[t]);
    }
  }
  const int nl = lane & 15, half = lane >> 4;
#pragma unroll
  for (int g = 0; g < 8; ++g) {
    const int orow = tile + half * 8 + g;
    const float cx = cx2in[orow];
    float y[NTILES];
#pragma unroll
    for (int t = 0; t < NTILES; ++t) y[t] = acc[t][g];
    const float cxn = poincare_row<NTILES, !FINAL>(y, cx, coef, ch, sh, nl);
    if constexpr (FINAL) {
      if (nl < NREAL) fout[(size_t)orow * NREAL + nl] = y[0];
    } else {
#pragma unroll
      for (int t = 0; t < NTILES; ++t)
        nxt[(size_t)orow * (NTILES * 16) + t * 16 + nl] = (_Float16)y[t];
      if (nl == 0) cx2out[orow] = cxn;
    }
  }
}

// ---------------------------------------------------------------------------
extern "C" void kernel_launch(void* const* d_in, const int* in_sizes, int n_in,
                              void* d_out, int out_size, void* d_ws, size_t ws_size,
                              hipStream_t stream) {
  (void)in_sizes; (void)n_in; (void)out_size; (void)ws_size;
  const float* x   = (const float*)d_in[0];
  const float* z1  = (const float*)d_in[1];
  const float* rb1 = (const float*)d_in[2];
  const float* z2  = (const float*)d_in[3];
  const float* rb2 = (const float*)d_in[4];
  const float* zF1 = (const float*)d_in[5];
  const float* rF1 = (const float*)d_in[6];
  const float* zF2 = (const float*)d_in[7];
  const float* rF2 = (const float*)d_in[8];
  const float* zF3 = (const float*)d_in[9];
  const float* rF3 = (const float*)d_in[10];
  float* out = (float*)d_out;
  char* ws = (char*)d_ws;

  // beta-concatenation ratios (host lgamma; baked into graph args)
  auto betaR = [](double n1, double n2) -> float {
    return (float)exp(lgamma(n1 * 0.5) - lgamma((n1 + 1.0) * 0.5)
                    - lgamma(n2 * 0.5) + lgamma((n2 + 1.0) * 0.5));
  };
  const float ratio1 = betaR(75.0, 3.0);
  const float ratio2 = betaR(150.0, 6.0);
  const float ratioF = betaR(400.0, 16.0);

  // --- workspace layout (bytes), regions reused across dead stages ---
  float*    par = (float*)(ws + 0);
  float *coL1 = par + 0,   *chL1 = par + 16,  *shL1 = par + 32;
  float *coL2 = par + 48,  *chL2 = par + 64,  *shL2 = par + 80;
  float *coF1 = par + 96,  *chF1 = par + 224, *shF1 = par + 352;
  float *coF2 = par + 480, *chF2 = par + 576, *shF2 = par + 672;
  float *coF3 = par + 768, *chF3 = par + 784, *shF3 = par + 800;
  _Float16* bf   = (_Float16*)(ws + 4096);
  _Float16 *wq1  = bf + 0,     *wq2  = bf + 1536,  *wqf1 = bf + 4096;
  _Float16 *wqf2 = bf + 57344, *wqf3 = bf + 69632;

  float*    vbuf = (float*)(ws + 262144);          // v1 (12.6MB) / v2
  float*    hbuf = (float*)(ws + 12845056);        // conv output h1 / h2
  float*    pbuf = (float*)(ws + 32112640);        // pooled2 (raw)
  _Float16* xf   = (_Float16*)(ws + 36929536);     // fc input [1024,416]
  float*    cxf  = (float*)(ws + 37781504);
  _Float16* hf1  = (_Float16*)(ws + 37785600);     // [1024,128]
  float*    cxf1 = (float*)(ws + 38047744);
  _Float16* hf2  = (_Float16*)(ws + 38051840);     // [1024,96]
  float*    cxf2 = (float*)(ws + 38248448);

  // --- pack all 5 weight layers into WMMA B-fragment layout ---
  pack_layer<<<6,   256, 0, stream>>>(z1,  rb1,  75,   6, 3, 1, coL1, chL1, shL1, wq1);
  pack_layer<<<10,  256, 0, stream>>>(z2,  rb2, 150,  16, 5, 1, coL2, chL2, shL2, wq2);
  pack_layer<<<208, 256, 0, stream>>>(zF1, rF1, 400, 120, 13, 8, coF1, chF1, shF1, wqf1);
  pack_layer<<<48,  256, 0, stream>>>(zF2, rF2, 120,  84, 4, 6, coF2, chF2, shF2, wqf2);
  pack_layer<<<6,   256, 0, stream>>>(zF3, rF3,  84,  10, 3, 1, coF3, chF3, shF3, wqf3);

  // --- stage 1: conv1 (implicit GEMM 802816 x 75 x 6, fully fused) ---
  logmap_prep<3><<<4096, 256, 0, stream>>>(x, vbuf, 1024, 1048576, ratio1);
  conv_gemm<3, 32, 28, 3, 75, 6><<<6272, 256, 0, stream>>>(
      vbuf, wq1, coL1, chL1, shL1, hbuf, 50176);
  // pool1 fused with stage-2 logmap: emits v2 directly
  hmaxpool<6, 28, true><<<784, 256, 0, stream>>>(hbuf, vbuf, 200704, ratio2);

  // --- stage 2: conv2 (102400 x 150 x 16, fully fused) ---
  conv_gemm<6, 14, 10, 5, 150, 16><<<800, 256, 0, stream>>>(
      vbuf, wq2, coL2, chL2, shL2, hbuf, 6400);
  hmaxpool<16, 10, false><<<100, 256, 0, stream>>>(hbuf, pbuf, 25600, 1.0f);

  // --- flatten + fc chain ---
  flatten_k<<<4, 256, 0, stream>>>(pbuf, xf, cxf, ratioF);
  fc_gemm<416, 13, 8, 120, false><<<8, 256, 0, stream>>>(
      xf, cxf, wqf1, coF1, chF1, shF1, hf1, cxf1, nullptr, 64);
  fc_gemm<128, 4, 6, 84, false><<<8, 256, 0, stream>>>(
      hf1, cxf1, wqf2, coF2, chF2, shF2, hf2, cxf2, nullptr, 64);
  fc_gemm<96, 3, 1, 10, true><<<8, 256, 0, stream>>>(
      hf2, cxf2, wqf3, coF3, chF3, shF3, nullptr, nullptr, out, 64);
}
